// GCN_30992484008171
// MI455X (gfx1250) — matrix operand order repaired and checked by reading, not compiled, not run on hardware
//
#include <hip/hip_runtime.h>
#include <stdint.h>
#include <stddef.h>

// ---------------- problem constants (match reference) ----------------
#define NNODES   100000
#define NEDGES   1600000
#define INFEATS  256
#define NHIDDEN  128
#define NCLASSES 40

// GEMM1: K=256 -> 8 ksteps of 32; N=128 -> 8 ntiles of 16
// GEMM2: K=128 -> 4 ksteps;        N=40  -> 3 ntiles (last partially valid)
#define KSTEPS1 8
#define NTILES1 8
#define KSTEPS2 4
#define NTILES2 3

typedef __attribute__((ext_vector_type(16))) __bf16 v16bf;
typedef __attribute__((ext_vector_type(8)))  float  v8f;

union FragU { v16bf v; uint4 q[2]; };   // 16 bf16 = 32 bytes = two uint4

__device__ __forceinline__ unsigned short f2bf(float f) {
  // round-to-nearest-even f32 -> bf16
  unsigned int u = __float_as_uint(f);
  u += 0x7FFFu + ((u >> 16) & 1u);
  return (unsigned short)(u >> 16);
}

// ---------------- small prep kernels ----------------

// deg_out[src] += 1 ; deg_in[dst] += 1
__global__ void gcn_degree(const int* __restrict__ esrc, const int* __restrict__ edst,
                           float* __restrict__ deg_out, float* __restrict__ deg_in, int nE) {
  int e = blockIdx.x * blockDim.x + threadIdx.x;
  if (e >= nE) return;
  atomicAdd(deg_out + esrc[e], 1.0f);
  atomicAdd(deg_in  + edst[e], 1.0f);
}

// norm = rsqrt(max(deg, 1))
__global__ void gcn_norms(const float* __restrict__ deg_out, const float* __restrict__ deg_in,
                          float* __restrict__ nsrc, float* __restrict__ ndst, int nN) {
  int i = blockIdx.x * blockDim.x + threadIdx.x;
  if (i >= nN) return;
  nsrc[i] = rsqrtf(fmaxf(deg_out[i], 1.0f));
  ndst[i] = rsqrtf(fmaxf(deg_in[i],  1.0f));
}

// f32 -> bf16 row-major copy (for X)
__global__ void gcn_cvt_bf16(const float* __restrict__ src, unsigned short* __restrict__ dst,
                             size_t n) {
  size_t i = (size_t)blockIdx.x * blockDim.x + threadIdx.x;
  if (i >= n) return;
  dst[i] = f2bf(src[i]);
}

// Pack a row-major (Ksrc x Nsrc) f32 weight matrix into per-lane B-fragment
// layout for v_wmma_f32_16x16x32_bf16:
//   Wp[((ntile*ksteps + kstep)*32 + lane)*16 + s]
//     = bf16( W[(kstep*32 + (s&7) + 8*(lane>=16) + 16*(s>=8)) * Nsrc + ntile*16 + (lane&15)] )
// (B 32x16: lanes 0-15 hold K=0..7,16..23, lanes 16-31 hold K=8..15,24..31; column = lane&15)
__global__ void gcn_packB(const float* __restrict__ W, unsigned short* __restrict__ Wp,
                          int Nsrc, int ksteps, int ntiles) {
  int idx = blockIdx.x * blockDim.x + threadIdx.x;
  int total = ntiles * ksteps * 32 * 16;
  if (idx >= total) return;
  int s    = idx & 15;
  int lane = (idx >> 4) & 31;
  int rest = idx >> 9;               // = ntile*ksteps + kstep
  int kstep = rest % ksteps;
  int ntile = rest / ksteps;
  int k   = kstep * 32 + (s & 7) + ((lane >> 4) << 3) + ((s >> 3) << 4);
  int col = ntile * 16 + (lane & 15);
  float v = (col < Nsrc) ? W[(size_t)k * Nsrc + col] : 0.0f;
  Wp[idx] = f2bf(v);
}

// ---------------- WMMA GEMM: Out[row,col] = (A @ B)[row,col] * nsrc[row] ----------------
// A: bf16 row-major (M x lda), lda = 32*ksteps. One wave per 16x16 tile.
// blockDim.x = 32 * ntiles ; gridDim.x = M/16 (M divisible by 16 here: 100000 = 6250*16).
__global__ void gcn_gemm_bf16(const unsigned short* __restrict__ A, int lda,
                              const unsigned short* __restrict__ Bp, int ksteps,
                              const float* __restrict__ nsrc,
                              float* __restrict__ Out, int ncols) {
  const int lane   = threadIdx.x & 31;
  const int ntile  = threadIdx.x >> 5;
  const int rowbase = blockIdx.x << 4;
  const int r  = lane & 15;
  const int hi = (lane >> 4) & 1;

  v8f acc = {};
  for (int ks = 0; ks < ksteps; ++ks) {
    // A fragment: row r, K = {k0..k0+7, k0+16..k0+23}, k0 = ks*32 + 8*hi
    const int k0 = ks * 32 + (hi << 3);
    const uint4* ap = (const uint4*)(A + (size_t)(rowbase + r) * lda + k0);
    FragU Af;
    Af.q[0] = ap[0];   // K = k0 .. k0+7
    Af.q[1] = ap[2];   // K = k0+16 .. k0+23  (+32 bytes)
    // B fragment: packed, one contiguous 32B load per lane
    const uint4* bp = (const uint4*)(Bp + ((size_t)(ntile * ksteps + ks) * 32 + lane) * 16);
    FragU Bf;
    Bf.q[0] = bp[0];
    Bf.q[1] = bp[1];
    acc = __builtin_amdgcn_wmma_f32_16x16x32_bf16(
        /*neg_a=*/false, Af.v, /*neg_b=*/false, Bf.v,
        /*c_mod=*/(short)0, acc, /*reuse_a=*/false, /*reuse_b=*/false);
  }

  // D layout: lane holds (m = v + 8*hi, n = lane&15) for v = 0..7
  const int col = (ntile << 4) + r;
  if (col < ncols) {
    #pragma unroll
    for (int v = 0; v < 8; ++v) {
      const int row = rowbase + v + (hi << 3);
      Out[(size_t)row * ncols + col] = acc[v] * nsrc[row];
    }
  }
}

// ---------------- edge aggregation: Agg[dst,:] += H[src,:] ----------------
// one block per edge, blockDim = feature count (rounded up to wave multiple)
__global__ void gcn_aggregate(const int* __restrict__ esrc, const int* __restrict__ edst,
                              const float* __restrict__ H, float* __restrict__ Agg,
                              int nf, int nE) {
  const int e = blockIdx.x;
  const int f = threadIdx.x;
  const int s = esrc[e];
  const int d = edst[e];
  if (f == 0 && e + 1 < nE) {
    // warm next edge's source row into cache (global_prefetch_b8)
    __builtin_prefetch(H + (size_t)esrc[e + 1] * nf, 0, 1);
  }
  if (f < nf) {
    atomicAdd(Agg + (size_t)d * nf + f, H[(size_t)s * nf + f]);
  }
}

// h1 = relu(agg1 * ndst + b1), stored as bf16 row-major (feeds GEMM2 A)
__global__ void gcn_finalize1(const float* __restrict__ agg, const float* __restrict__ ndst,
                              const float* __restrict__ b1, unsigned short* __restrict__ h1b,
                              size_t total) {
  size_t i = (size_t)blockIdx.x * blockDim.x + threadIdx.x;
  if (i >= total) return;
  int n = (int)(i >> 7);        // / 128
  int f = (int)(i & 127);
  float v = agg[i] * ndst[n] + b1[f];
  h1b[i] = f2bf(fmaxf(v, 0.0f));
}

// out = agg2 * ndst + b2
__global__ void gcn_finalize2(const float* __restrict__ agg, const float* __restrict__ ndst,
                              const float* __restrict__ b2, float* __restrict__ out,
                              size_t total) {
  size_t i = (size_t)blockIdx.x * blockDim.x + threadIdx.x;
  if (i >= total) return;
  int n = (int)(i / NCLASSES);
  int f = (int)(i % NCLASSES);
  out[i] = agg[i] * ndst[n] + b2[f];
}

// ---------------- launcher ----------------
extern "C" void kernel_launch(void* const* d_in, const int* in_sizes, int n_in,
                              void* d_out, int out_size, void* d_ws, size_t ws_size,
                              hipStream_t stream) {
  const float* X    = (const float*)d_in[0];
  const int*   esrc = (const int*)  d_in[1];
  const int*   edst = (const int*)  d_in[2];
  const float* W1   = (const float*)d_in[3];
  const float* b1   = (const float*)d_in[4];
  const float* W2   = (const float*)d_in[5];
  const float* b2   = (const float*)d_in[6];
  float* out = (float*)d_out;

  const int nN = in_sizes[0] / INFEATS;   // 100000
  const int nE = in_sizes[1];             // 1600000

  // carve workspace (256B aligned chunks)
  char* p = (char*)d_ws;
  auto carve = [&](size_t bytes) -> void* {
    void* r = (void*)p;
    p += (bytes + 255) & ~(size_t)255;
    return r;
  };
  float* deg_out = (float*)carve((size_t)nN * 4);
  float* deg_in  = (float*)carve((size_t)nN * 4);
  float* nsrc    = (float*)carve((size_t)nN * 4);
  float* ndst    = (float*)carve((size_t)nN * 4);
  unsigned short* Xb  = (unsigned short*)carve((size_t)nN * INFEATS * 2);
  unsigned short* W1p = (unsigned short*)carve((size_t)NTILES1 * KSTEPS1 * 32 * 16 * 2);
  unsigned short* W2p = (unsigned short*)carve((size_t)NTILES2 * KSTEPS2 * 32 * 16 * 2);
  float* hpre  = (float*)carve((size_t)nN * NHIDDEN * 4);
  float* agg1  = (float*)carve((size_t)nN * NHIDDEN * 4);
  unsigned short* h1b = (unsigned short*)carve((size_t)nN * NHIDDEN * 2);
  float* h2pre = (float*)carve((size_t)nN * NCLASSES * 4);
  float* agg2  = (float*)carve((size_t)nN * NCLASSES * 4);

  // zero the accumulators every call (workspace is not re-poisoned between replays)
  hipMemsetAsync(deg_out, 0, (size_t)nN * 4, stream);
  hipMemsetAsync(deg_in,  0, (size_t)nN * 4, stream);
  hipMemsetAsync(agg1,    0, (size_t)nN * NHIDDEN * 4, stream);
  hipMemsetAsync(agg2,    0, (size_t)nN * NCLASSES * 4, stream);

  // degrees + norms
  gcn_degree<<<(nE + 255) / 256, 256, 0, stream>>>(esrc, edst, deg_out, deg_in, nE);
  gcn_norms<<<(nN + 255) / 256, 256, 0, stream>>>(deg_out, deg_in, nsrc, ndst, nN);

  // convert X to bf16, pack weights into B-fragment layout
  {
    size_t total = (size_t)nN * INFEATS;
    gcn_cvt_bf16<<<(unsigned)((total + 255) / 256), 256, 0, stream>>>(X, Xb, total);
  }
  gcn_packB<<<(NTILES1 * KSTEPS1 * 32 * 16 + 255) / 256, 256, 0, stream>>>(
      W1, W1p, NHIDDEN, KSTEPS1, NTILES1);
  gcn_packB<<<(NTILES2 * KSTEPS2 * 32 * 16 + 255) / 256, 256, 0, stream>>>(
      W2, W2p, NCLASSES, KSTEPS2, NTILES2);

  const int mtiles = nN / 16;   // 6250 (nN divisible by 16)

  // layer 1: hpre = (X @ W1) * nsrc   (WMMA bf16, f32 accum)
  gcn_gemm_bf16<<<mtiles, 32 * NTILES1, 0, stream>>>(Xb, INFEATS, W1p, KSTEPS1,
                                                     nsrc, hpre, NHIDDEN);
  // agg1[dst,:] += hpre[src,:]
  gcn_aggregate<<<nE, NHIDDEN, 0, stream>>>(esrc, edst, hpre, agg1, NHIDDEN, nE);
  // h1 = relu(agg1 * ndst + b1) -> bf16
  {
    size_t total = (size_t)nN * NHIDDEN;
    gcn_finalize1<<<(unsigned)((total + 255) / 256), 256, 0, stream>>>(agg1, ndst, b1, h1b, total);
  }

  // layer 2: h2pre = (h1 @ W2) * nsrc
  gcn_gemm_bf16<<<mtiles, 32 * NTILES2, 0, stream>>>(h1b, NHIDDEN, W2p, KSTEPS2,
                                                     nsrc, h2pre, NCLASSES);
  // agg2[dst,:] += h2pre[src,:]   (block of 64 covers 40 feats, guard inside)
  gcn_aggregate<<<nE, 64, 0, stream>>>(esrc, edst, h2pre, agg2, NCLASSES, nE);
  // out = agg2 * ndst + b2
  {
    size_t total = (size_t)nN * NCLASSES;
    gcn_finalize2<<<(unsigned)((total + 255) / 256), 256, 0, stream>>>(agg2, ndst, b2, out, total);
  }
}
